// GAT_LSTM_50259707298004
// MI455X (gfx1250) — matrix operand order repaired and checked by reading, not compiled, not run on hardware
//
#include <hip/hip_runtime.h>
#include <hip/hip_fp16.h>
#include <math.h>

typedef _Float16 f16;
typedef __attribute__((ext_vector_type(16))) _Float16 v16h;
typedef __attribute__((ext_vector_type(8)))  float    v8f;
typedef __attribute__((ext_vector_type(4)))  float    v4f;   // clang native vec for NT loads

#define N_NODES 156
#define T_STEPS 2048
#define NFEAT_  256
#define NHID_   128
#define HID     8
#define G4      32      /* 4*H gate rows */
#define KPAD    160     /* 156 padded to 5*32 for WMMA K-steps */
#define ALPHA_  0.2f

__device__ __forceinline__ float fast_tanhf(float x) {
#if __has_builtin(__builtin_amdgcn_tanhf)
    return __builtin_amdgcn_tanhf(x);
#else
    return tanhf(x);
#endif
}

// ---------------------------------------------------------------------------
// Kernel 1 (tiny, 1 block): attention scores -> softmax A -> M = W_ih @ A
// (f16, K-padded), plus fused bias b_ih+b_hh.
// ---------------------------------------------------------------------------
__global__ __launch_bounds__(256) void k_prep(
    const float* __restrict__ emb, const float* __restrict__ W,
    const float* __restrict__ a,   const float* __restrict__ W_ih,
    const float* __restrict__ b_ih,const float* __restrict__ b_hh,
    float* __restrict__ A, f16* __restrict__ Mh, float* __restrict__ bsum)
{
    __shared__ float w1[NFEAT_], w2[NFEAT_];
    __shared__ float s1[N_NODES], s2[N_NODES];
    const int tid = threadIdx.x;

    // w1 = W @ a[:128], w2 = W @ a[128:]
    {
        const float* wr = W + tid * NHID_;
        float a1 = 0.f, a2 = 0.f;
        for (int m = 0; m < NHID_; ++m) {
            float wv = wr[m];
            a1 += wv * a[m];
            a2 += wv * a[NHID_ + m];
        }
        w1[tid] = a1; w2[tid] = a2;
    }
    __syncthreads();

    if (tid < N_NODES) {
        const float* er = emb + tid * NFEAT_;
        float a1 = 0.f, a2 = 0.f;
        for (int f = 0; f < NFEAT_; ++f) { float e = er[f]; a1 += e * w1[f]; a2 += e * w2[f]; }
        s1[tid] = a1; s2[tid] = a2;
    }
    __syncthreads();

    if (tid < N_NODES) {  // row-wise leaky-relu + softmax over j
        float si = s1[tid];
        float mx = -1e30f;
        for (int j = 0; j < N_NODES; ++j) {
            float e = si + s2[j]; e = e > 0.f ? e : ALPHA_ * e;
            mx = fmaxf(mx, e);
        }
        float sum = 0.f;
        for (int j = 0; j < N_NODES; ++j) {
            float e = si + s2[j]; e = e > 0.f ? e : ALPHA_ * e;
            sum += __expf(e - mx);
        }
        float inv = 1.f / sum;
        float* Ar = A + tid * N_NODES;
        for (int j = 0; j < N_NODES; ++j) {
            float e = si + s2[j]; e = e > 0.f ? e : ALPHA_ * e;
            Ar[j] = __expf(e - mx) * inv;
        }
    }
    if (tid < G4) bsum[tid] = b_ih[tid] + b_hh[tid];
    __threadfence();
    __syncthreads();

    // M[g,j] = sum_k W_ih[g,k] * A[k,j]; store f16, zero-padded j -> KPAD
    for (int idx = tid; idx < G4 * KPAD; idx += 256) {
        int g = idx / KPAD, j = idx % KPAD;
        float acc = 0.f;
        if (j < N_NODES) {
            const float* wr = W_ih + g * N_NODES;
            for (int k = 0; k < N_NODES; ++k) acc += wr[k] * A[k * N_NODES + j];
        }
        Mh[idx] = (f16)acc;
    }
}

// ---------------------------------------------------------------------------
// Kernel 2: G[(n*T+t), g] = sum_j x[n,t,j] * M[g,j] + b[g]
// One wave per 16-row tile; 5 K-steps x 2 WMMAs (v_wmma_f32_16x16x32_f16).
// B fragments (M) live in VGPRs for the whole kernel.  x is streamed with
// non-temporal loads (read-once, 199 MB) so the 41 MB G buffer stays
// resident in the 192 MB L2 for the downstream LSTM scan.
// ---------------------------------------------------------------------------
__global__ __launch_bounds__(256) void k_gemm(
    const float* __restrict__ X, const f16* __restrict__ Mh,
    const float* __restrict__ bsum, float* __restrict__ G)
{
    const int lane   = threadIdx.x & 31;
    const int wave   = blockIdx.x * (blockDim.x >> 5) + (threadIdx.x >> 5);
    const int nWaves = gridDim.x * (blockDim.x >> 5);
    const int col    = lane & 15;
    const bool hiH   = lane >= 16;

    // B fragment layout (16-bit 32x16): lanes 0-15 -> N=lane, K=0..15;
    // lanes 16-31 -> N=lane-16, K=16..31.  M rows are contiguous in j.
    v16h Bf[5][2];
#pragma unroll
    for (int ks = 0; ks < 5; ++ks) {
#pragma unroll
        for (int gt = 0; gt < 2; ++gt) {
            int kb = ks * 32 + (hiH ? 16 : 0);
            Bf[ks][gt] = *(const v16h*)(Mh + (size_t)(gt * 16 + col) * KPAD + kb);
        }
    }
    const float bias0 = bsum[col];
    const float bias1 = bsum[16 + col];

    const int nTiles = (N_NODES * T_STEPS) / 16;   // 19968
    for (int tile = wave; tile < nTiles; tile += nWaves) {
        const float* rp = X + (size_t)(tile * 16 + col) * N_NODES;
        v8f acc0 = {}, acc1 = {};
#pragma unroll
        for (int ks = 0; ks < 5; ++ks) {
            // A fragment (16-bit 16x32): lanes 0-15: halves 0..15=K0..15;
            // lanes 16-31: halves 0..15=K16..31 (matches B).
            const int jA = ks * 32 + (hiH ? 16 : 0);
            const int jB = jA + 8;
            v16h af;
            if (ks < 4) {
                v4f f0 = __builtin_nontemporal_load((const v4f*)(rp + jA));
                v4f f1 = __builtin_nontemporal_load((const v4f*)(rp + jA + 4));
                v4f f2 = __builtin_nontemporal_load((const v4f*)(rp + jB));
                v4f f3 = __builtin_nontemporal_load((const v4f*)(rp + jB + 4));
                af[0]=(f16)f0.x; af[1]=(f16)f0.y; af[2] =(f16)f0.z; af[3] =(f16)f0.w;
                af[4]=(f16)f1.x; af[5]=(f16)f1.y; af[6] =(f16)f1.z; af[7] =(f16)f1.w;
                af[8]=(f16)f2.x; af[9]=(f16)f2.y; af[10]=(f16)f2.z; af[11]=(f16)f2.w;
                af[12]=(f16)f3.x;af[13]=(f16)f3.y;af[14]=(f16)f3.z; af[15]=(f16)f3.w;
            } else {                                // K tail: j in [128,160) -> guard >=156
#pragma unroll
                for (int h = 0; h < 8; ++h) {
                    int j1 = jA + h, j2 = jB + h;
                    int c1 = j1 < N_NODES ? j1 : 0, c2 = j2 < N_NODES ? j2 : 0;
                    float v1 = __builtin_nontemporal_load(rp + c1);
                    float v2 = __builtin_nontemporal_load(rp + c2);
                    af[h]     = (f16)(j1 < N_NODES ? v1 : 0.f);
                    af[8 + h] = (f16)(j2 < N_NODES ? v2 : 0.f);
                }
            }
            acc0 = __builtin_amdgcn_wmma_f32_16x16x32_f16(false, af, false, Bf[ks][0],
                                                          (short)0, acc0, false, false);
            acc1 = __builtin_amdgcn_wmma_f32_16x16x32_f16(false, af, false, Bf[ks][1],
                                                          (short)0, acc1, false, false);
        }
        // D layout: VGPR v -> row (v | v+8), N = lane&15
        float* gp = G + (size_t)tile * 16 * G4;
        const int mofs = hiH ? 8 : 0;
#pragma unroll
        for (int v = 0; v < 8; ++v) {
            gp[(v + mofs) * G4 + col]      = acc0[v] + bias0;
            gp[(v + mofs) * G4 + 16 + col] = acc1[v] + bias1;
        }
    }
}

// ---------------------------------------------------------------------------
// Kernel 3: 156 independent LSTMs, one wave32 per node; lane = gate row.
// sigma(x) = 0.5 + 0.5*tanh(x/2) -> one v_tanh per lane for the gate activation.
// G is read from L2 (kept resident by NT policy on x).  Final FC fused.
// ---------------------------------------------------------------------------
__global__ __launch_bounds__(128) void k_lstm(
    const float* __restrict__ G, const float* __restrict__ W_hh,
    const float* __restrict__ W_fc, const float* __restrict__ b_fc,
    float* __restrict__ out)
{
    const int lane = threadIdx.x & 31;
    const int node = blockIdx.x * (blockDim.x >> 5) + (threadIdx.x >> 5);
    if (node >= N_NODES) return;

    const float* Gp = G + (size_t)node * T_STEPS * G4;
    const float* wr = W_hh + lane * HID;
    const float w0 = wr[0], w1 = wr[1], w2 = wr[2], w3 = wr[3];
    const float w4 = wr[4], w5 = wr[5], w6 = wr[6], w7 = wr[7];
    const int grp = lane >> 3;      // 0:i 1:f 2:g 3:o
    const int idx = lane & 7;

    float hp0=0.f,hp1=0.f,hp2=0.f,hp3=0.f,hp4=0.f,hp5=0.f,hp6=0.f,hp7=0.f;
    float c = 0.f;
    float gv = Gp[lane];

    for (int t = 0; t < T_STEPS; ++t) {
        int tn = t + 1; if (tn >= T_STEPS) tn = T_STEPS - 1;
        float gnext = Gp[(size_t)tn * G4 + lane];        // prefetch next step

        float g = gv + w0*hp0 + w1*hp1 + w2*hp2 + w3*hp3
                     + w4*hp4 + w5*hp5 + w6*hp6 + w7*hp7;
        float sg  = (grp == 2) ? g : 0.5f * g;
        float th  = fast_tanhf(sg);
        float act = (grp == 2) ? th : 0.5f + 0.5f * th;  // tanh or sigmoid

        float i_  = __shfl(act, idx,       32);
        float f_  = __shfl(act, idx + 8,   32);
        float gc_ = __shfl(act, idx + 16,  32);
        float o_  = __shfl(act, idx + 24,  32);
        c = f_ * c + i_ * gc_;
        float hn = o_ * fast_tanhf(c);

        hp0 = __shfl(hn, 0, 32); hp1 = __shfl(hn, 1, 32);
        hp2 = __shfl(hn, 2, 32); hp3 = __shfl(hn, 3, 32);
        hp4 = __shfl(hn, 4, 32); hp5 = __shfl(hn, 5, 32);
        hp6 = __shfl(hn, 6, 32); hp7 = __shfl(hn, 7, 32);
        gv = gnext;
    }

    // out[n,m] = hT[n,:] . W_fc[m,:] + b_fc[m]
    for (int m = lane; m < N_NODES; m += 32) {
        const float* fr = W_fc + m * HID;
        float o = b_fc[m] + hp0*fr[0] + hp1*fr[1] + hp2*fr[2] + hp3*fr[3]
                          + hp4*fr[4] + hp5*fr[5] + hp6*fr[6] + hp7*fr[7];
        out[node * N_NODES + m] = o;
    }
}

// ---------------------------------------------------------------------------
extern "C" void kernel_launch(void* const* d_in, const int* in_sizes, int n_in,
                              void* d_out, int out_size, void* d_ws, size_t ws_size,
                              hipStream_t stream) {
    const float* emb  = (const float*)d_in[0];
    const float* x    = (const float*)d_in[1];
    /* d_in[2] = adj: unused by the reference math */
    const float* W    = (const float*)d_in[3];
    const float* a    = (const float*)d_in[4];
    const float* W_ih = (const float*)d_in[5];
    const float* W_hh = (const float*)d_in[6];
    const float* b_ih = (const float*)d_in[7];
    const float* b_hh = (const float*)d_in[8];
    const float* W_fc = (const float*)d_in[9];
    const float* b_fc = (const float*)d_in[10];
    float* out = (float*)d_out;

    char* ws = (char*)d_ws;
    float* A    = (float*)(ws);            // 156*156*4     = 97,344 B
    f16*   Mh   = (f16*)  (ws + 98304);    // 32*160*2      = 10,240 B
    float* bsum = (float*)(ws + 110592);   // 32*4
    float* G    = (float*)(ws + 131072);   // 156*2048*32*4 = 40,894,464 B

    k_prep<<<1,    256, 0, stream>>>(emb, W, a, W_ih, b_ih, b_hh, A, Mh, bsum);
    k_gemm<<<1248, 256, 0, stream>>>(x, Mh, bsum, G);   // 9984 waves, 2 tiles each
    k_lstm<<<39,   128, 0, stream>>>(G, W_hh, W_fc, b_fc, out);
}
